// ActorAC_81509889344041
// MI455X (gfx1250) — compile-verified, matrix operand order checked by reading, not run on hardware
//
#include <hip/hip_runtime.h>
#include <hip/hip_bf16.h>
#include <math.h>

typedef __attribute__((ext_vector_type(16))) _Float16 v16h;
typedef __attribute__((ext_vector_type(8)))  _Float16 v8h;
typedef __attribute__((ext_vector_type(8)))  float    v8f;

#define NTOK 64
#define DIN  2
#define DQ   8
#define DK   8
#define DV   16
#define DH   128
#define DOUT 10

// ---------------------------------------------------------------------------
// Pack a row-major f32 weight matrix W[Kdim][Ndim] into WMMA-B-fragment order
// (f16).  For tile (kt, nt) each lane's 16 elements are contiguous:
//   frag base = ((kt*NT + nt)*32 + lane)*16
//   lane l: col = nt*16 + (l&15), kbase = kt*32 + (l>>4)*8
//   elem i<8  -> K = kbase + i
//   elem i>=8 -> K = kbase + 16 + (i-8)
// Out-of-range K/col are zero-padded (handles Kdim=16 pad->32, Ndim=10 pad->16).
// ---------------------------------------------------------------------------
__global__ void pack_wmma_b_kernel(const float* __restrict__ W,
                                   _Float16* __restrict__ dst,
                                   int Kdim, int Ndim, int KT, int NT) {
    int total = KT * NT * 512;
    int e = blockIdx.x * blockDim.x + threadIdx.x;
    if (e >= total) return;
    int i    = e & 15;
    int lane = (e >> 4) & 31;
    int tile = e >> 9;
    int nt = tile % NT;
    int kt = tile / NT;
    int ncol  = nt * 16 + (lane & 15);
    int kbase = kt * 32 + (lane >> 4) * 8;
    int k = (i < 8) ? (kbase + i) : (kbase + 16 + (i - 8));
    float v = (k < Kdim && ncol < Ndim) ? W[k * Ndim + ncol] : 0.0f;
    dst[e] = (_Float16)v;
}

__device__ __forceinline__ v16h frag_combine(v8h lo, v8h hi) {
    v16h r;
#pragma unroll
    for (int i = 0; i < 8; ++i) { r[i] = lo[i]; r[8 + i] = hi[i]; }
    return r;
}

__device__ __forceinline__ v16h load_a_frag(const _Float16* rowPtr) {
    // rowPtr points at element K = kbase of this lane's activation row.
    v8h lo = *(const v8h*)(rowPtr);
    v8h hi = *(const v8h*)(rowPtr + 16);
    return frag_combine(lo, hi);
}

__device__ __forceinline__ v16h load_b_frag(const _Float16* p) {
    const v8h* bp = (const v8h*)p;
    return frag_combine(bp[0], bp[1]);
}

// Branch-free tanh: tanh(x) = 1 - 2/(exp(2x)+1).
// Clamp keeps exp finite (|x|<=9 -> exp(18) ~ 6.6e7, exact saturation anyway).
// Lowers to: min/max, mul, v_exp_f32 (via __expf), add, v_rcp_f32, fma.
__device__ __forceinline__ float fast_tanh(float x) {
    float cx = fminf(fmaxf(x, -9.0f), 9.0f);
    float e2 = __expf(2.0f * cx);
    float r  = __builtin_amdgcn_rcpf(e2 + 1.0f);
    return fmaf(-2.0f, r, 1.0f);
}

// ---------------------------------------------------------------------------
// One block per batch element. 256 threads = 8 waves (wave32).
// ---------------------------------------------------------------------------
__global__ __launch_bounds__(256)
void actor_attn_mlp_kernel(const float* __restrict__ state,
                           const float* __restrict__ WQ, const float* __restrict__ bQ,
                           const float* __restrict__ WK, const float* __restrict__ bK,
                           const float* __restrict__ WV, const float* __restrict__ bV,
                           const _Float16* __restrict__ pW1o,
                           const _Float16* __restrict__ pW2o,
                           const _Float16* __restrict__ pW3o,
                           const float* __restrict__ b1o, const float* __restrict__ b2o,
                           const float* __restrict__ b3o,
                           const _Float16* __restrict__ pW1r,
                           const _Float16* __restrict__ pW2r,
                           const _Float16* __restrict__ pW3r,
                           const float* __restrict__ b1r, const float* __restrict__ b2r,
                           const float* __restrict__ b3r,
                           float* __restrict__ out, int nBatch) {
    __shared__ float    s_q[NTOK][DQ];
    __shared__ float    s_k[NTOK][DK];
    __shared__ float    s_v[NTOK][DV];
    __shared__ float    s_attn[NTOK][NTOK];
    __shared__ _Float16 s_ctx[NTOK][32];     // DV=16 padded to K=32 for WMMA
    __shared__ _Float16 s_h[NTOK][DH];
    __shared__ _Float16 s_h2[NTOK][DH];

    const int b    = blockIdx.x;
    const int tid  = threadIdx.x;
    const int lane = tid & 31;
    const int wave = tid >> 5;
    const int half = lane >> 4;
    const int lrow = lane & 15;

    // ---- Phase 1: Q/K/V projections (DIN=2, trivial) --------------------
    if (tid < NTOK) {
        const float s0 = state[(size_t)b * NTOK * DIN + tid * DIN + 0];
        const float s1 = state[(size_t)b * NTOK * DIN + tid * DIN + 1];
#pragma unroll
        for (int j = 0; j < DQ; ++j)
            s_q[tid][j] = fmaf(s0, WQ[j], fmaf(s1, WQ[DQ + j], bQ[j]));
#pragma unroll
        for (int j = 0; j < DK; ++j)
            s_k[tid][j] = fmaf(s0, WK[j], fmaf(s1, WK[DK + j], bK[j]));
#pragma unroll
        for (int j = 0; j < DV; ++j)
            s_v[tid][j] = fmaf(s0, WV[j], fmaf(s1, WV[DV + j], bV[j]));
    }
    __syncthreads();

    // ---- Phase 2a: scores = Q K^T / sqrt(8), 256 threads -> (row, quad) --
    {
        const int n = tid & 63;
        const int q = tid >> 6;
        const float scale = 0.35355339059327373f; // 1/sqrt(8)
#pragma unroll
        for (int mm = 0; mm < 16; ++mm) {
            const int m = q * 16 + mm;
            float d = 0.0f;
#pragma unroll
            for (int k = 0; k < DK; ++k) d = fmaf(s_q[n][k], s_k[m][k], d);
            s_attn[n][m] = d * scale;
        }
    }
    __syncthreads();

    // ---- Phase 2b: softmax per row (64 threads) --------------------------
    if (tid < NTOK) {
        float mx = -INFINITY;
#pragma unroll 8
        for (int m = 0; m < NTOK; ++m) mx = fmaxf(mx, s_attn[tid][m]);
        float sum = 0.0f;
#pragma unroll 8
        for (int m = 0; m < NTOK; ++m) {
            const float e = __expf(s_attn[tid][m] - mx);
            s_attn[tid][m] = e;
            sum += e;
        }
        const float inv = __builtin_amdgcn_rcpf(sum);
#pragma unroll 8
        for (int m = 0; m < NTOK; ++m) s_attn[tid][m] *= inv;
    }
    __syncthreads();

    // ---- Phase 3: ctx = attn @ V -> f16, K padded to 32 ------------------
    {
        const int n = tid >> 2;      // token row
        const int g = tid & 3;       // 4 columns per thread
#pragma unroll
        for (int dd = 0; dd < 4; ++dd) {
            const int d = g * 4 + dd;
            float acc = 0.0f;
#pragma unroll 8
            for (int m = 0; m < NTOK; ++m) acc = fmaf(s_attn[n][m], s_v[m][d], acc);
            s_ctx[n][d] = (_Float16)acc;
            s_ctx[n][16 + d] = (_Float16)0.0f;   // zero-pad K 16..31
        }
    }
    __syncthreads();

    // ---- Phase 4: dual 3-layer tanh MLP heads via WMMA -------------------
    const _Float16* pw1[2] = {pW1o, pW1r};
    const _Float16* pw2[2] = {pW2o, pW2r};
    const _Float16* pw3[2] = {pW3o, pW3r};
    const float* pb1[2] = {b1o, b1r};
    const float* pb2[2] = {b2o, b2r};
    const float* pb3[2] = {b3o, b3r};

#pragma unroll
    for (int head = 0; head < 2; ++head) {
        // Layer 1: h = tanh(ctx @ W1 + b1)   M=64,K=32(1 tile),N=128
        // wave w owns n-tile w; loop 4 m-tiles.
        {
            const v16h bfrag = load_b_frag(pw1[head] + ((size_t)wave * 512 + lane * 16));
            const float bv = pb1[head][wave * 16 + lrow];
#pragma unroll
            for (int mt = 0; mt < 4; ++mt) {
                v8f acc = {};
                v16h a = load_a_frag(&s_ctx[mt * 16 + lrow][half * 8]);
                acc = __builtin_amdgcn_wmma_f32_16x16x32_f16(
                    false, a, false, bfrag, (short)0, acc, false, false);
#pragma unroll
                for (int i = 0; i < 8; ++i)
                    s_h[mt * 16 + i + 8 * half][wave * 16 + lrow] =
                        (_Float16)fast_tanh(acc[i] + bv);
            }
        }
        __syncthreads();

        // Layer 2: h2 = tanh(h @ W2 + b2)    M=64,K=128(4 tiles),N=128
        {
            const float bv = pb2[head][wave * 16 + lrow];
#pragma unroll
            for (int mt = 0; mt < 4; ++mt) {
                v8f acc = {};
#pragma unroll
                for (int kt = 0; kt < 4; ++kt) {
                    v16h a = load_a_frag(&s_h[mt * 16 + lrow][kt * 32 + half * 8]);
                    v16h bf = load_b_frag(pw2[head] +
                        ((size_t)(kt * 8 + wave) * 512 + lane * 16));
                    acc = __builtin_amdgcn_wmma_f32_16x16x32_f16(
                        false, a, false, bf, (short)0, acc, false, false);
                }
#pragma unroll
                for (int i = 0; i < 8; ++i)
                    s_h2[mt * 16 + i + 8 * half][wave * 16 + lrow] =
                        (_Float16)fast_tanh(acc[i] + bv);
            }
        }
        __syncthreads();

        // Layer 3: out = h2 @ W3 + b3        M=64,K=128(4 tiles),N=10 (pad 16)
        if (wave < 4) {
            const int mt = wave;
            v8f acc = {};
#pragma unroll
            for (int kt = 0; kt < 4; ++kt) {
                v16h a = load_a_frag(&s_h2[mt * 16 + lrow][kt * 32 + half * 8]);
                v16h bf = load_b_frag(pw3[head] + ((size_t)kt * 512 + lane * 16));
                acc = __builtin_amdgcn_wmma_f32_16x16x32_f16(
                    false, a, false, bf, (short)0, acc, false, false);
            }
            if (lrow < DOUT) {
                const float bv = pb3[head][lrow];
                float* outh = out + (size_t)head * nBatch * NTOK * DOUT
                                  + (size_t)b * NTOK * DOUT;
#pragma unroll
                for (int i = 0; i < 8; ++i) {
                    const int tok = mt * 16 + i + 8 * half;
                    outh[tok * DOUT + lrow] = acc[i] + bv;
                }
            }
        }
        __syncthreads();   // before next head reuses s_h / s_h2
    }
}

extern "C" void kernel_launch(void* const* d_in, const int* in_sizes, int n_in,
                              void* d_out, int out_size, void* d_ws, size_t ws_size,
                              hipStream_t stream) {
    const float* state = (const float*)d_in[0];
    const float* WQ  = (const float*)d_in[1];
    const float* bQ  = (const float*)d_in[2];
    const float* WK  = (const float*)d_in[3];
    const float* bK  = (const float*)d_in[4];
    const float* WV  = (const float*)d_in[5];
    const float* bV  = (const float*)d_in[6];
    const float* W1o = (const float*)d_in[7];
    const float* b1o = (const float*)d_in[8];
    const float* W2o = (const float*)d_in[9];
    const float* b2o = (const float*)d_in[10];
    const float* W3o = (const float*)d_in[11];
    const float* b3o = (const float*)d_in[12];
    const float* W1r = (const float*)d_in[13];
    const float* b1r = (const float*)d_in[14];
    const float* W2r = (const float*)d_in[15];
    const float* b2r = (const float*)d_in[16];
    const float* W3r = (const float*)d_in[17];
    const float* b3r = (const float*)d_in[18];
    float* out = (float*)d_out;

    const int nBatch = in_sizes[0] / (NTOK * DIN);   // 4096

    // Packed f16 weight fragments in workspace.
    // W1: KT=1,NT=8 -> 4096 halves ; W2: KT=4,NT=8 -> 16384 ; W3: KT=4,NT=1 -> 2048
    _Float16* ws = (_Float16*)d_ws;
    _Float16* pW1o = ws + 0;
    _Float16* pW2o = ws + 4096;
    _Float16* pW3o = ws + 20480;
    _Float16* pW1r = ws + 22528;
    _Float16* pW2r = ws + 26624;
    _Float16* pW3r = ws + 43008;

    pack_wmma_b_kernel<<<(4096 + 255) / 256, 256, 0, stream>>>(W1o, pW1o, DV, DH, 1, 8);
    pack_wmma_b_kernel<<<(16384 + 255) / 256, 256, 0, stream>>>(W2o, pW2o, DH, DH, 4, 8);
    pack_wmma_b_kernel<<<(2048 + 255) / 256, 256, 0, stream>>>(W3o, pW3o, DH, DOUT, 4, 1);
    pack_wmma_b_kernel<<<(4096 + 255) / 256, 256, 0, stream>>>(W1r, pW1r, DV, DH, 1, 8);
    pack_wmma_b_kernel<<<(16384 + 255) / 256, 256, 0, stream>>>(W2r, pW2r, DH, DH, 4, 8);
    pack_wmma_b_kernel<<<(2048 + 255) / 256, 256, 0, stream>>>(W3r, pW3r, DH, DOUT, 4, 1);

    actor_attn_mlp_kernel<<<nBatch, 256, 0, stream>>>(
        state, WQ, bQ, WK, bK, WV, bV,
        pW1o, pW2o, pW3o, b1o, b2o, b3o,
        pW1r, pW2r, pW3r, b1r, b2r, b3r,
        out, nBatch);
}
